// MoE_40372692582952
// MI455X (gfx1250) — compile-verified, matrix operand order checked by reading, not run on hardware
//
#include <hip/hip_runtime.h>

// ---------------- types ----------------
typedef __bf16 bf16_t;
typedef __attribute__((ext_vector_type(16))) __bf16 v16bf;
typedef __attribute__((ext_vector_type(8)))  float  v8f;

// ---------------- problem constants ----------------
#define TOK   2048      // B*S
#define DDIM  768       // D
#define IDIM  384       // I
#define T2I   768       // 2*I
#define NEXP  9         // 8 routed + 1 shared
#define RCAP  6656      // 4608 (padded routed rows) + 2048 (shared rows)

#define MT 64           // block tile M
#define NTB 128         // block tile N
#define KT 32           // K step (bf16 WMMA K=32)

// =====================================================================
// prepack: f32 -> bf16, weights transposed to [expert][N][K] (K contiguous)
// =====================================================================
__global__ void pack_x_kernel(const float* __restrict__ x, bf16_t* __restrict__ xbf, int n) {
  int i = blockIdx.x * 256 + threadIdx.x;
  if (i < n) xbf[i] = (bf16_t)x[i];
}

__global__ void pack_w13_kernel(const float* __restrict__ W1, const float* __restrict__ Ws1,
                                const float* __restrict__ W3, const float* __restrict__ Ws3,
                                bf16_t* __restrict__ w1p, bf16_t* __restrict__ w3p, int n) {
  int i = blockIdx.x * 256 + threadIdx.x;
  if (i >= n) return;
  int e    = i / (T2I * DDIM);
  int rem  = i - e * (T2I * DDIM);
  int ncol = rem / DDIM;          // column in 2I space (N)
  int d    = rem - ncol * DDIM;   // K index
  float v1, v3;
  if (e < 8) {
    size_t s = (size_t)e * DDIM * T2I + (size_t)d * T2I + ncol;  // W[e][d][n]
    v1 = W1[s]; v3 = W3[s];
  } else {
    size_t s = (size_t)d * T2I + ncol;
    v1 = Ws1[s]; v3 = Ws3[s];
  }
  w1p[i] = (bf16_t)v1;
  w3p[i] = (bf16_t)v3;
}

__global__ void pack_w2_kernel(const float* __restrict__ W2, const float* __restrict__ Ws2,
                               bf16_t* __restrict__ w2p, int n) {
  int i = blockIdx.x * 256 + threadIdx.x;
  if (i >= n) return;
  int e   = i / (DDIM * IDIM);
  int rem = i - e * (DDIM * IDIM);
  int d   = rem / IDIM;           // N (output col)
  int k   = rem - d * IDIM;       // K
  float v;
  if (e < 8) v = W2[(size_t)e * IDIM * DDIM + (size_t)k * DDIM + d];  // W2[e][k][d]
  else       v = Ws2[(size_t)k * DDIM + d];
  w2p[i] = (bf16_t)v;
}

// =====================================================================
// router: one wave per token, scalar f32 (routing is precision-sensitive)
// =====================================================================
__global__ __launch_bounds__(256)
void gate_kernel(const float* __restrict__ x, const float* __restrict__ gw,
                 const float* __restrict__ gb,
                 int* __restrict__ topidx, float* __restrict__ topw) {
  int t = blockIdx.x * 8 + (threadIdx.x >> 5);
  if (t >= TOK) return;
  int lane = threadIdx.x & 31;
  float acc[8];
#pragma unroll
  for (int e = 0; e < 8; ++e) acc[e] = 0.f;
  const float* xr = x + (size_t)t * DDIM;
  for (int d = lane; d < DDIM; d += 32) {
    float xv = xr[d];
#pragma unroll
    for (int e = 0; e < 8; ++e) acc[e] += xv * gw[e * DDIM + d];
  }
#pragma unroll
  for (int e = 0; e < 8; ++e)
    for (int off = 16; off; off >>= 1) acc[e] += __shfl_xor(acc[e], off, 32);
  if (lane != 0) return;

  float m = acc[0];
  for (int e = 1; e < 8; ++e) m = fmaxf(m, acc[e]);
  float p[8], sum = 0.f;
  for (int e = 0; e < 8; ++e) { p[e] = __expf(acc[e] - m); sum += p[e]; }
  float sc[8], s[8];
  for (int e = 0; e < 8; ++e) { sc[e] = p[e] / sum; s[e] = sc[e] + gb[e]; }
  // group scores: G=4 groups of 2, top-2-of-2 == sum
  float gsc[4];
  for (int g = 0; g < 4; ++g) gsc[g] = s[2 * g] + s[2 * g + 1];
  int g0 = 0;
  for (int g = 1; g < 4; ++g) if (gsc[g] > gsc[g0]) g0 = g;
  int g1 = -1;
  for (int g = 0; g < 4; ++g) { if (g == g0) continue; if (g1 < 0 || gsc[g] > gsc[g1]) g1 = g; }
  float ms[8];
  for (int e = 0; e < 8; ++e) { int g = e >> 1; ms[e] = (g == g0 || g == g1) ? s[e] : -3.4e38f; }
  int i0 = 0;
  for (int e = 1; e < 8; ++e) if (ms[e] > ms[i0]) i0 = e;
  int i1 = -1;
  for (int e = 0; e < 8; ++e) { if (e == i0) continue; if (i1 < 0 || ms[e] > ms[i1]) i1 = e; }
  topidx[t * 2 + 0] = i0;  topidx[t * 2 + 1] = i1;
  topw[t * 2 + 0] = sc[i0]; topw[t * 2 + 1] = sc[i1];
}

// =====================================================================
// deterministic list builder: per-expert token lists, 64-aligned segments
// seg[0..8]=base, seg[9..17]=count
// =====================================================================
__global__ void build_lists_kernel(const int* __restrict__ topidx,
                                   int* __restrict__ rowTok,
                                   int* __restrict__ tokpos,
                                   int* __restrict__ seg) {
  __shared__ int cnt[8];
  __shared__ int base[9];
  int tid = threadIdx.x;
  if (tid < 8) {
    int c = 0;
    for (int t = 0; t < TOK; ++t) {
      if (topidx[t * 2 + 0] == tid) c++;
      if (topidx[t * 2 + 1] == tid) c++;
    }
    cnt[tid] = c;
  }
  __syncthreads();
  if (tid == 0) {
    int b = 0;
    for (int e = 0; e < 8; ++e) { base[e] = b; b += (cnt[e] + 63) & ~63; }
    base[8] = b;
    for (int e = 0; e < 8; ++e) { seg[e] = base[e]; seg[9 + e] = cnt[e]; }
    seg[8] = b; seg[9 + 8] = TOK;
  }
  __syncthreads();
  if (tid < 8) {
    int pos = base[tid];
    for (int t = 0; t < TOK; ++t)
      for (int k = 0; k < 2; ++k)
        if (topidx[t * 2 + k] == tid) { rowTok[pos] = t; tokpos[t * 2 + k] = pos; pos++; }
    int end = base[tid] + ((cnt[tid] + 63) & ~63);
    for (int r = pos; r < end; ++r) rowTok[r] = -1;
  } else if (tid == 8) {
    int b8 = base[8];
    for (int t = 0; t < TOK; ++t) rowTok[b8 + t] = t;
  }
}

// =====================================================================
// GEMM1: gathered X(bf16) @ [W1|W3](bf16, [e][N][K]) -> fused SwiGLU -> Hg(bf16)
// block: 256 thr = 8 waves; block tile 64M x 128N, K-step 32.
// wave (wm 0..1, wn 0..3): 32M x 32N register tile, both W1 and W3:
//   2 A-frags x 2 N-subtiles x {W1,W3} = 8 WMMAs / K-step.
// LDS A tile double-buffered: one barrier per K-step, gathered global loads
// for step k+1 overlap the WMMAs of step k.
// =====================================================================
__global__ __launch_bounds__(256)
void gemm1_kernel(const bf16_t* __restrict__ xbf,
                  const bf16_t* __restrict__ w1p,
                  const bf16_t* __restrict__ w3p,
                  const float* __restrict__ B1, const float* __restrict__ Bs1,
                  const float* __restrict__ B3, const float* __restrict__ Bs3,
                  const int* __restrict__ rowTok,
                  const int* __restrict__ seg,
                  bf16_t* __restrict__ hg) {
  const int e = blockIdx.z;
  const int segBase = seg[e];
  const int segCnt  = seg[9 + e];
  const int m0 = blockIdx.x * MT;
  if (m0 >= segCnt) return;
  const int n0 = blockIdx.y * NTB;

  __shared__ __attribute__((aligned(32))) bf16_t sA[2][MT][KT];
  __shared__ int sTok[MT];

  const int tid  = threadIdx.x;
  const int lane = tid & 31;
  const int wave = tid >> 5;
  const int wm = wave & 1;          // 0..1 -> 32 rows each
  const int wn = wave >> 1;         // 0..3 -> 32 h-cols each

  if (tid < MT) {
    int r = m0 + tid;
    sTok[tid] = (r < segCnt) ? rowTok[segBase + r] : -1;
  }
  __syncthreads();

  const bf16_t* wb1 = w1p + (size_t)e * T2I * DDIM;
  const bf16_t* wb3 = w3p + (size_t)e * T2I * DDIM;

  v8f acc1[2][2] = {};   // [mSub][nSub] for W1
  v8f acc3[2][2] = {};   // [mSub][nSub] for W3

  const int aRow  = lane & 15;
  const int aK    = (lane >> 4) * 16;
  const int bCol0 = n0 + wn * 32 + (lane & 15);
  const int bKOff = (lane >> 4) * 16;

  const int sr = tid >> 2;          // staging row 0..63
  const int sc = (tid & 3) * 8;     // staging col 0/8/16/24
  const int tokR = sTok[sr];
  const bf16_t* aG = (tokR >= 0) ? (xbf + (size_t)tokR * DDIM + sc) : nullptr;

  // prologue: stage K-step 0
  {
    uint4 v = make_uint4(0, 0, 0, 0);
    if (tokR >= 0) v = *(const uint4*)(aG + 0);
    *(uint4*)&sA[0][sr][sc] = v;
  }
  __syncthreads();

  const int NIT = DDIM / KT;        // 24
  for (int it = 0; it < NIT; ++it) {
    const int cur = it & 1;
    const int kk  = it * KT;
    // kick off next tile's gathered global loads (overlaps WMMAs below)
    uint4 nv = make_uint4(0, 0, 0, 0);
    if (it + 1 < NIT && tokR >= 0) nv = *(const uint4*)(aG + (it + 1) * KT);

    v16bf a0 = *(const v16bf*)&sA[cur][wm * 32 + aRow][aK];
    v16bf a1 = *(const v16bf*)&sA[cur][wm * 32 + 16 + aRow][aK];
    v16bf b1A = *(const v16bf*)(wb1 + (size_t)bCol0 * DDIM + kk + bKOff);
    v16bf b1B = *(const v16bf*)(wb1 + (size_t)(bCol0 + 16) * DDIM + kk + bKOff);
    v16bf b3A = *(const v16bf*)(wb3 + (size_t)bCol0 * DDIM + kk + bKOff);
    v16bf b3B = *(const v16bf*)(wb3 + (size_t)(bCol0 + 16) * DDIM + kk + bKOff);

    acc1[0][0] = __builtin_amdgcn_wmma_f32_16x16x32_bf16(false, a0, false, b1A, (short)0, acc1[0][0], false, false);
    acc1[0][1] = __builtin_amdgcn_wmma_f32_16x16x32_bf16(false, a0, false, b1B, (short)0, acc1[0][1], false, false);
    acc1[1][0] = __builtin_amdgcn_wmma_f32_16x16x32_bf16(false, a1, false, b1A, (short)0, acc1[1][0], false, false);
    acc1[1][1] = __builtin_amdgcn_wmma_f32_16x16x32_bf16(false, a1, false, b1B, (short)0, acc1[1][1], false, false);
    acc3[0][0] = __builtin_amdgcn_wmma_f32_16x16x32_bf16(false, a0, false, b3A, (short)0, acc3[0][0], false, false);
    acc3[0][1] = __builtin_amdgcn_wmma_f32_16x16x32_bf16(false, a0, false, b3B, (short)0, acc3[0][1], false, false);
    acc3[1][0] = __builtin_amdgcn_wmma_f32_16x16x32_bf16(false, a1, false, b3A, (short)0, acc3[1][0], false, false);
    acc3[1][1] = __builtin_amdgcn_wmma_f32_16x16x32_bf16(false, a1, false, b3B, (short)0, acc3[1][1], false, false);

    if (it + 1 < NIT) *(uint4*)&sA[1 - cur][sr][sc] = nv;
    __syncthreads();
  }

  const float* b1p = (e < 8) ? (B1 + e * T2I) : Bs1;
  const float* b3p = (e < 8) ? (B3 + e * T2I) : Bs3;
  const int laneHi = (lane >> 4) << 3;   // M offset 0 or 8 in C layout

#pragma unroll
  for (int ms = 0; ms < 2; ++ms) {
#pragma unroll
    for (int ns = 0; ns < 2; ++ns) {
      v8f a1 = acc1[ms][ns];
      v8f a3 = acc3[ms][ns];
      int col = n0 + wn * 32 + ns * 16 + (lane & 15);
      float bb1 = b1p[col];
      float bb3 = b3p[col];
#pragma unroll
      for (int v = 0; v < 8; ++v) {
        int mloc = wm * 32 + ms * 16 + v + laneHi;
        float h  = (a1[v] + bb1) * (a3[v] + bb3);
        float hn = __shfl_xor(h, 1, 32);      // partner (odd/even) column
        float xg = fminf(h, 7.0f);
        float xl = fminf(fmaxf(hn, -7.0f), 7.0f);
        float g  = xg * (1.0f / (1.0f + __expf(-1.702f * xg))) * (xl + 1.0f);
        int rowg = m0 + mloc;
        if (((lane & 1) == 0) && rowg < segCnt)
          hg[(size_t)(segBase + rowg) * IDIM + (col >> 1)] = (bf16_t)g;
      }
    }
  }
}

// =====================================================================
// GEMM2: Hg(bf16, gathered order) @ W2(bf16, [e][D][I]) + B2 -> y (f32)
// block tile 64M x 128N, wave 32M x 32N: 2 A-frags x 2 B-frags = 4 WMMAs/step
// =====================================================================
__global__ __launch_bounds__(256)
void gemm2_kernel(const bf16_t* __restrict__ hg,
                  const bf16_t* __restrict__ w2p,
                  const float* __restrict__ B2, const float* __restrict__ Bs2,
                  const int* __restrict__ seg,
                  float* __restrict__ y) {
  const int e = blockIdx.z;
  const int segBase = seg[e];
  const int segCnt  = seg[9 + e];
  const int m0 = blockIdx.x * MT;
  if (m0 >= segCnt) return;
  const int n0 = blockIdx.y * NTB;

  __shared__ __attribute__((aligned(32))) bf16_t sA[2][MT][KT];

  const int tid  = threadIdx.x;
  const int lane = tid & 31;
  const int wave = tid >> 5;
  const int wm = wave & 1;
  const int wn = wave >> 1;

  const bf16_t* wb = w2p + (size_t)e * DDIM * IDIM;

  v8f acc[2][2] = {};    // [mSub][nSub]
  const int aRow  = lane & 15;
  const int aK    = (lane >> 4) * 16;
  const int bCol0 = n0 + wn * 32 + (lane & 15);
  const int bKOff = (lane >> 4) * 16;
  const int sr = tid >> 2;
  const int sc = (tid & 3) * 8;
  const bf16_t* aG = hg + (size_t)(segBase + m0 + sr) * IDIM + sc;

  *(uint4*)&sA[0][sr][sc] = *(const uint4*)(aG + 0);
  __syncthreads();

  const int NIT = IDIM / KT;        // 12
  for (int it = 0; it < NIT; ++it) {
    const int cur = it & 1;
    const int kk  = it * KT;
    uint4 nv = make_uint4(0, 0, 0, 0);
    if (it + 1 < NIT) nv = *(const uint4*)(aG + (it + 1) * KT);

    v16bf a0 = *(const v16bf*)&sA[cur][wm * 32 + aRow][aK];
    v16bf a1 = *(const v16bf*)&sA[cur][wm * 32 + 16 + aRow][aK];
    v16bf bA = *(const v16bf*)(wb + (size_t)bCol0 * IDIM + kk + bKOff);
    v16bf bB = *(const v16bf*)(wb + (size_t)(bCol0 + 16) * IDIM + kk + bKOff);

    acc[0][0] = __builtin_amdgcn_wmma_f32_16x16x32_bf16(false, a0, false, bA, (short)0, acc[0][0], false, false);
    acc[0][1] = __builtin_amdgcn_wmma_f32_16x16x32_bf16(false, a0, false, bB, (short)0, acc[0][1], false, false);
    acc[1][0] = __builtin_amdgcn_wmma_f32_16x16x32_bf16(false, a1, false, bA, (short)0, acc[1][0], false, false);
    acc[1][1] = __builtin_amdgcn_wmma_f32_16x16x32_bf16(false, a1, false, bB, (short)0, acc[1][1], false, false);

    if (it + 1 < NIT) *(uint4*)&sA[1 - cur][sr][sc] = nv;
    __syncthreads();
  }

  const float* b2p = (e < 8) ? (B2 + e * DDIM) : Bs2;
  const int laneHi = (lane >> 4) << 3;
#pragma unroll
  for (int ms = 0; ms < 2; ++ms) {
#pragma unroll
    for (int ns = 0; ns < 2; ++ns) {
      v8f a = acc[ms][ns];
      int col = n0 + wn * 32 + ns * 16 + (lane & 15);
      float bb = b2p[col];
#pragma unroll
      for (int v = 0; v < 8; ++v) {
        int mloc = wm * 32 + ms * 16 + v + laneHi;
        y[(size_t)(segBase + m0 + mloc) * DDIM + col] = a[v] + bb;
      }
    }
  }
}

// =====================================================================
// combine: out = y_shared + w0*y[pos0] + w1*y[pos1]  (fixed order -> deterministic)
// =====================================================================
__global__ void combine_kernel(const float* __restrict__ y,
                               const float* __restrict__ topw,
                               const int* __restrict__ tokpos,
                               const int* __restrict__ seg,
                               float* __restrict__ out, int n) {
  int i = blockIdx.x * 256 + threadIdx.x;
  if (i >= n) return;
  int t = i / DDIM;
  int d = i - t * DDIM;
  int sb = seg[8];
  float v = y[(size_t)(sb + t) * DDIM + d];
  int p0 = tokpos[t * 2 + 0], p1 = tokpos[t * 2 + 1];
  v += topw[t * 2 + 0] * y[(size_t)p0 * DDIM + d];
  v += topw[t * 2 + 1] * y[(size_t)p1 * DDIM + d];
  out[i] = v;
}

// =====================================================================
// host launcher
// =====================================================================
static inline size_t alignup(size_t v, size_t a) { return (v + a - 1) & ~(a - 1); }

extern "C" void kernel_launch(void* const* d_in, const int* in_sizes, int n_in,
                              void* d_out, int out_size, void* d_ws, size_t ws_size,
                              hipStream_t stream) {
  const float* x      = (const float*)d_in[0];
  const float* gate_w = (const float*)d_in[1];
  const float* gate_b = (const float*)d_in[2];
  const float* W1  = (const float*)d_in[3];
  const float* B1  = (const float*)d_in[4];
  const float* W3  = (const float*)d_in[5];
  const float* B3  = (const float*)d_in[6];
  const float* W2  = (const float*)d_in[7];
  const float* B2  = (const float*)d_in[8];
  const float* Ws1 = (const float*)d_in[9];
  const float* Bs1 = (const float*)d_in[10];
  const float* Ws3 = (const float*)d_in[11];
  const float* Bs3 = (const float*)d_in[12];
  const float* Ws2 = (const float*)d_in[13];
  const float* Bs2 = (const float*)d_in[14];

  char* w = (char*)d_ws;
  size_t off = 0;
  auto carve = [&](size_t bytes) -> char* {
    char* p = w + off;
    off = alignup(off + bytes, 256);
    return p;
  };
  bf16_t* xbf  = (bf16_t*)carve((size_t)TOK * DDIM * 2);
  bf16_t* w1p  = (bf16_t*)carve((size_t)NEXP * T2I * DDIM * 2);
  bf16_t* w3p  = (bf16_t*)carve((size_t)NEXP * T2I * DDIM * 2);
  bf16_t* w2p  = (bf16_t*)carve((size_t)NEXP * DDIM * IDIM * 2);
  bf16_t* hg   = (bf16_t*)carve((size_t)RCAP * IDIM * 2);
  float*  ybuf = (float*) carve((size_t)RCAP * DDIM * 4);
  int* rowTok  = (int*)   carve((size_t)RCAP * 4);
  int* tokpos  = (int*)   carve((size_t)TOK * 2 * 4);
  int* topidx  = (int*)   carve((size_t)TOK * 2 * 4);
  float* topw  = (float*) carve((size_t)TOK * 2 * 4);
  int* seg     = (int*)   carve(32 * 4);
  (void)ws_size; (void)n_in; (void)in_sizes; (void)out_size;

  // 1. precision prepack (f32 -> bf16, weights transposed to K-contiguous)
  {
    int n = TOK * DDIM;
    pack_x_kernel<<<(n + 255) / 256, 256, 0, stream>>>(x, xbf, n);
  }
  {
    int n = NEXP * T2I * DDIM;
    pack_w13_kernel<<<(n + 255) / 256, 256, 0, stream>>>(W1, Ws1, W3, Ws3, w1p, w3p, n);
  }
  {
    int n = NEXP * DDIM * IDIM;
    pack_w2_kernel<<<(n + 255) / 256, 256, 0, stream>>>(W2, Ws2, w2p, n);
  }

  // 2. router (f32 scalar)
  gate_kernel<<<TOK / 8, 256, 0, stream>>>(x, gate_w, gate_b, topidx, topw);

  // 3. deterministic token-list build
  build_lists_kernel<<<1, 32, 0, stream>>>(topidx, rowTok, tokpos, seg);

  // 4. grouped gathered GEMM1 (+ fused bias / h1*h3 / interleaved SwiGLU)
  gemm1_kernel<<<dim3(TOK / MT, T2I / NTB, NEXP), 256, 0, stream>>>(
      xbf, w1p, w3p, B1, Bs1, B3, Bs3, rowTok, seg, hg);

  // 5. grouped GEMM2 (+ bias) -> per-row f32 outputs
  gemm2_kernel<<<dim3(TOK / MT, DDIM / NTB, NEXP), 256, 0, stream>>>(
      hg, w2p, B2, Bs2, seg, ybuf);

  // 6. deterministic combine: shared + weighted routed
  {
    int n = TOK * DDIM;
    combine_kernel<<<(n + 255) / 256, 256, 0, stream>>>(ybuf, topw, tokpos, seg,
                                                        (float*)d_out, n);
  }
}
